// RotarySelfAttention_44710609551622
// MI455X (gfx1250) — matrix operand order repaired and checked by reading, not compiled, not run on hardware
//
#include <hip/hip_runtime.h>
#include <hip/hip_bf16.h>

// ---------------------------------------------------------------------------
// RotarySelfAttention (ragged KV-cache append + RoPE + SDPA) for gfx1250.
// Matmuls via V_WMMA_F32_16X16X32_F16; GEMM tiles staged with the Tensor
// Data Mover (tensor_load_to_lds + TENSORcnt) when the builtin is available.
// ---------------------------------------------------------------------------

typedef _Float16 half_t;
typedef __attribute__((ext_vector_type(16))) _Float16 v16h;
typedef __attribute__((ext_vector_type(8)))  float    v8f;
typedef __attribute__((ext_vector_type(4)))  unsigned int u32x4;
typedef __attribute__((ext_vector_type(8)))  int          i32x8;
typedef __attribute__((ext_vector_type(4)))  int          i32x4;

#define BSZ    8
#define TNEW   512
#define DMODEL 1024
#define NHEAD  16
#define DHEAD  64
#define NCACHE 2048
#define TTOT   2560            // NCACHE + TNEW
#define MROWS  (BSZ * TNEW)    // 4096

#if defined(__has_builtin)
#  if __has_builtin(__builtin_amdgcn_tensor_load_to_lds)
#    define HAVE_TDM 1
#  endif
#endif
#ifndef HAVE_TDM
#  define HAVE_TDM 0
#endif

// ---- WMMA fragment lane maps (CDNA5 ISA §7.12.2) ----------------
// A 16x32 f16: lane l holds row (l&15); half (l>>4) selects K subset.
__device__ __forceinline__ int frag_ak(int j, int lhalf) {
  int p = j >> 1;
  int base = (p < 4) ? (2 * p) : (16 + 2 * (p - 4));
  return base + 8 * lhalf + (j & 1);
}
// B 32x16 f16: same K order as A (k = 8*half + j for j<8, 16 + 8*half + (j-8) for j>=8).
__device__ __forceinline__ int frag_bk(int j, int lhalf) {
  return (j < 8) ? (j + 8 * lhalf) : (16 + 8 * lhalf + (j - 8));
}

__device__ __forceinline__ v8f wmma16(v16h a, v16h b, v8f c) {
  // (neg_a, A, neg_b, B, c_mod, C, reuse_a, reuse_b)
  return __builtin_amdgcn_wmma_f32_16x16x32_f16(false, a, false, b, (short)0, c,
                                                false, false);
}

#define LDSS 40  // LDS row stride in halves = 64B tile row + 16B TDM pad

#if HAVE_TDM
// ---------------------------------------------------------------------------
// Issue one TDM descriptor: 2-D f16 tile, 32 elements wide x 128 rows, from
// global (row stride = strideElems) into LDS at lds_addr, with 16B padding
// appended every 64B row (pad_interval=16 DWORDs, pad_amount=4 DWORDs) so the
// LDS image has a 40-half row stride (bank-conflict friendly, 16B aligned).
// D# field packing per CDNA5 ISA.
// ---------------------------------------------------------------------------
__device__ __forceinline__ void tdm_load_tile(unsigned lds_addr,
                                              const half_t* gptr,
                                              int strideElems, int nrows) {
  unsigned long long ga = (unsigned long long)(uintptr_t)gptr;
  u32x4 g0;
  g0[0] = 1u;                                   // count=1, user descriptor
  g0[1] = lds_addr;                             // LDS byte address
  g0[2] = (unsigned)(ga & 0xFFFFFFFFu);         // global_addr[31:0]
  g0[3] = (unsigned)((ga >> 32) & 0x1FFFFFFu)   // global_addr[56:32]
          | (2u << 30);                         // type = 2 ("image")
  i32x8 g1;
  g1[0] = (1 << 16)      // data_size = 2 bytes
        | (1 << 20)      // pad_enable
        | (3 << 22)      // pad_interval: 16 DWORDs (64B) between pads
        | (3 << 25);     // pad_amount:   4 DWORDs (16B)
  g1[1] = (int)(((unsigned)strideElems & 0xFFFFu) << 16);          // dim0 lo16
  g1[2] = (int)((((unsigned)strideElems >> 16) & 0xFFFFu)
        | (((unsigned)nrows & 0xFFFFu) << 16));                    // dim0 hi / dim1 lo
  g1[3] = (int)((((unsigned)nrows >> 16) & 0xFFFFu) | (32u << 16)); // dim1 hi, tile_dim0=32
  g1[4] = 128;                                   // tile_dim1=128, tile_dim2=0
  g1[5] = strideElems;                           // tensor_dim0_stride[31:0]
  g1[6] = 0;                                     // stride0 hi / stride1 lo
  g1[7] = 0;
  i32x4 z4 = {0, 0, 0, 0};
#if __clang_major__ >= 23
  i32x8 z8 = {0, 0, 0, 0, 0, 0, 0, 0};
  __builtin_amdgcn_tensor_load_to_lds(g0, g1, z4, z4, z8, 0);
#else
  __builtin_amdgcn_tensor_load_to_lds(g0, g1, z4, z4, 0);
#endif
}
#endif  // HAVE_TDM

// ---------------------------------------------------------------------------
// fp32 -> fp16 conversion (grid stride)
// ---------------------------------------------------------------------------
__global__ void k_f32_to_f16(const float* __restrict__ src,
                             half_t* __restrict__ dst, long n) {
  long i = (long)blockIdx.x * blockDim.x + threadIdx.x;
  long stride = (long)gridDim.x * blockDim.x;
  for (; i < n; i += stride) dst[i] = (half_t)src[i];
}

// ---------------------------------------------------------------------------
// total_lengths = past_lengths + sum(valid_new_mask)
// ---------------------------------------------------------------------------
__global__ void k_lens(const unsigned char* __restrict__ valid,
                       const int* __restrict__ plen,
                       int* __restrict__ lens_ws, int* __restrict__ lens_out) {
  int b = threadIdx.x;
  if (b < BSZ) {
    int c = 0;
    for (int t = 0; t < TNEW; ++t) c += (valid[b * TNEW + t] ? 1 : 0);
    int tot = plen[b] + c;
    lens_ws[b]  = tot;
    lens_out[b] = tot;
  }
}

// ---------------------------------------------------------------------------
// C[M,N] = A[M,K] @ W[N,K]^T + bias  (optionally row-masked), f16 in / f32 out
// Block tile 128x128, 8 waves, each wave: 64(M) x 32(N) via 4x2 WMMA tiles.
// TDM path: wave 0 double-buffers both tiles through the Tensor Data Mover
// while all waves compute; sync via s_wait_tensorcnt + workgroup barrier.
// ---------------------------------------------------------------------------
__global__ __launch_bounds__(256)
void k_gemm_wT(const half_t* __restrict__ A, const half_t* __restrict__ W,
               const float* __restrict__ bias, float* __restrict__ C,
               int M, int N, int K, const unsigned char* __restrict__ rowmask) {
#if HAVE_TDM
  __shared__ __align__(16) half_t As[2][128][LDSS];
  __shared__ __align__(16) half_t Bs[2][128][LDSS];
#else
  __shared__ __align__(16) half_t As[1][128][LDSS];
  __shared__ __align__(16) half_t Bs[1][128][LDSS];
#endif

  const int tid   = threadIdx.x;
  const int lane  = tid & 31;
  const int wave  = tid >> 5;
  const int lhalf = lane >> 4;
  const int l15   = lane & 15;
  const int wm    = wave & 1;   // 0..1 -> 64 rows each
  const int wn    = wave >> 1;  // 0..3 -> 32 cols each
  const int m0    = blockIdx.x * 128;
  const int n0    = blockIdx.y * 128;

  v8f acc[4][2] = {};

#if HAVE_TDM
  const half_t* Abase = A + (size_t)m0 * K;
  const half_t* Bbase = W + (size_t)n0 * K;
  const unsigned ldsA0 = (unsigned)(uintptr_t)&As[0][0][0];
  const unsigned ldsA1 = (unsigned)(uintptr_t)&As[1][0][0];
  const unsigned ldsB0 = (unsigned)(uintptr_t)&Bs[0][0][0];
  const unsigned ldsB1 = (unsigned)(uintptr_t)&Bs[1][0][0];

  if (wave == 0) {  // prologue: tile 0 -> buffer 0
    tdm_load_tile(ldsA0, Abase, K, M);
    tdm_load_tile(ldsB0, Bbase, K, N);
  }
  for (int k0 = 0; k0 < K; k0 += 32) {
    const int buf = (k0 >> 5) & 1;
    if (wave == 0) {
      if (k0 + 32 < K) {  // issue next tile into the other buffer
        tdm_load_tile(buf ? ldsA0 : ldsA1, Abase + k0 + 32, K, M);
        tdm_load_tile(buf ? ldsB0 : ldsB1, Bbase + k0 + 32, K, N);
        // TDM is in-order per wave: <=2 outstanding means tile k0 is resident
        __builtin_amdgcn_s_wait_tensorcnt(2);
      } else {
        __builtin_amdgcn_s_wait_tensorcnt(0);
      }
    }
    __syncthreads();  // buffer `buf` now visible to all waves
#else
  const int ldr = tid >> 1;        // 0..127 : tile row loaded by this thread
  const int ldc = (tid & 1) * 16;  // half-row segment (16 halves)
  for (int k0 = 0; k0 < K; k0 += 32) {
    const int buf = 0;
    __syncthreads();
    const uint4* ga =
        reinterpret_cast<const uint4*>(A + (size_t)(m0 + ldr) * K + k0 + ldc);
    const uint4* gb =
        reinterpret_cast<const uint4*>(W + (size_t)(n0 + ldr) * K + k0 + ldc);
    uint4 a0 = ga[0], a1 = ga[1];
    uint4 b0 = gb[0], b1 = gb[1];
    if (k0 + 32 < K) {  // pull next K-slab toward GL2 (global_prefetch_b8)
      __builtin_prefetch(A + (size_t)(m0 + ldr) * K + k0 + 32 + ldc, 0, 1);
      __builtin_prefetch(W + (size_t)(n0 + ldr) * K + k0 + 32 + ldc, 0, 1);
    }
    *reinterpret_cast<uint4*>(&As[0][ldr][ldc])     = a0;
    *reinterpret_cast<uint4*>(&As[0][ldr][ldc + 8]) = a1;
    *reinterpret_cast<uint4*>(&Bs[0][ldr][ldc])     = b0;
    *reinterpret_cast<uint4*>(&Bs[0][ldr][ldc + 8]) = b1;
    __syncthreads();
#endif

    v16h af[4];
#pragma unroll
    for (int fm = 0; fm < 4; ++fm) {
      int row = wm * 64 + fm * 16 + l15;
#pragma unroll
      for (int j = 0; j < 16; ++j) af[fm][j] = As[buf][row][frag_ak(j, lhalf)];
    }
    v16h bf[2];
#pragma unroll
    for (int fn = 0; fn < 2; ++fn) {
      int col = wn * 32 + fn * 16 + l15;
#pragma unroll
      for (int j = 0; j < 16; ++j) bf[fn][j] = Bs[buf][col][frag_bk(j, lhalf)];
    }
#pragma unroll
    for (int fm = 0; fm < 4; ++fm)
#pragma unroll
      for (int fn = 0; fn < 2; ++fn)
        acc[fm][fn] = wmma16(af[fm], bf[fn], acc[fm][fn]);

#if HAVE_TDM
    __syncthreads();  // everyone done reading `buf` before TDM reuses it
#endif
  }

#pragma unroll
  for (int fm = 0; fm < 4; ++fm)
#pragma unroll
    for (int fn = 0; fn < 2; ++fn)
#pragma unroll
      for (int r = 0; r < 8; ++r) {
        int m = m0 + wm * 64 + fm * 16 + r + 8 * lhalf;
        int n = n0 + wn * 32 + fn * 16 + l15;
        float v = acc[fm][fn][r] + bias[n];
        if (rowmask) v *= (float)rowmask[m];
        C[(size_t)m * N + n] = v;
      }
}

// ---------------------------------------------------------------------------
// Initialize k_total / v_total: copy past (t < past_len) else 0; write fp32
// result tensors and f16 shadow copies for WMMA attention.
// ---------------------------------------------------------------------------
__global__ void k_init_past(const float* __restrict__ pk,
                            const float* __restrict__ pv,
                            const int* __restrict__ plen,
                            float* __restrict__ kT, float* __restrict__ vT,
                            half_t* __restrict__ khT, half_t* __restrict__ vhT) {
  const long n = (long)BSZ * NHEAD * TTOT * DHEAD;
  long i = (long)blockIdx.x * blockDim.x + threadIdx.x;
  long stride = (long)gridDim.x * blockDim.x;
  for (; i < n; i += stride) {
    int d  = (int)(i & (DHEAD - 1));
    int t  = (int)((i >> 6) % TTOT);
    int bh = (int)(i / ((long)DHEAD * TTOT));
    int b  = bh >> 4;
    float kv = 0.f, vv = 0.f;
    if (t < plen[b]) {
      size_t pidx = (((size_t)bh) * NCACHE + t) * DHEAD + d;
      kv = pk[pidx];
      vv = pv[pidx];
    }
    kT[i]  = kv;
    vT[i]  = vv;
    khT[i] = (half_t)kv;
    vhT[i] = (half_t)vv;
  }
}

// ---------------------------------------------------------------------------
// RoPE on q / k_new, mask invalid tokens, ragged scatter of k/v into cache.
// grid (TNEW, NHEAD, BSZ), 64 threads = one d each.
// ---------------------------------------------------------------------------
__global__ __launch_bounds__(64)
void k_rope_scatter(const float* __restrict__ qf, const float* __restrict__ kf,
                    const float* __restrict__ vf,
                    const float* __restrict__ inv_freq,
                    const int* __restrict__ plen,
                    const unsigned char* __restrict__ valid,
                    half_t* __restrict__ qh, float* __restrict__ kT,
                    float* __restrict__ vT, half_t* __restrict__ khT,
                    half_t* __restrict__ vhT) {
  const int d = threadIdx.x;
  const int t = blockIdx.x, h = blockIdx.y, b = blockIdx.z;
  const size_t row = ((size_t)b * TNEW + t) * DMODEL + h * DHEAD;
  const int isval = valid[b * TNEW + t] ? 1 : 0;
  const int pl    = plen[b];

  float q = qf[row + d], k = kf[row + d], v = vf[row + d];
  float pos = isval ? (float)(pl + t) : 0.f;
  float fr  = inv_freq[d & 31] * pos;
  float cs  = __cosf(fr), sn = __sinf(fr);
  float qo  = (d < 32) ? qf[row + d + 32] : qf[row + d - 32];
  float ko  = (d < 32) ? kf[row + d + 32] : kf[row + d - 32];
  float sgn = (d < 32) ? -1.f : 1.f;
  float qr = q * cs + sgn * qo * sn;
  float kr = k * cs + sgn * ko * sn;

  size_t qidx = (((size_t)(b * NHEAD + h)) * TNEW + t) * DHEAD + d;
  qh[qidx] = (half_t)(isval ? qr : 0.f);

  if (isval) {
    int p = pl + t;
    size_t kidx = (((size_t)(b * NHEAD + h)) * TTOT + p) * DHEAD + d;
    kT[kidx]  = kr;
    vT[kidx]  = v;
    khT[kidx] = (half_t)kr;
    vhT[kidx] = (half_t)v;
  }
}

// ---------------------------------------------------------------------------
// Flash attention: block = (b, h, 64-row q tile); 4 waves x 16 q-rows.
// Per 32-key block: S = QK^T (4 WMMA), online softmax, ctx += P V (4 WMMA).
// ---------------------------------------------------------------------------
__global__ __launch_bounds__(128)
void k_attn(const half_t* __restrict__ qh, const half_t* __restrict__ khT,
            const half_t* __restrict__ vhT, const int* __restrict__ lens,
            half_t* __restrict__ ctxh) {
  __shared__ __align__(16) half_t Sbuf[4][16][32];  // wave-private P staging

  const int tid = threadIdx.x, lane = tid & 31, wave = tid >> 5;
  const int lhalf = lane >> 4, l15 = lane & 15;
  const int bid = blockIdx.x;
  const int qt = bid & 7, h = (bid >> 3) & (NHEAD - 1), b = bid >> 7;
  const int t0 = qt * 64 + wave * 16;
  const int tlen = lens[b];

  const half_t* Qb = qh  + ((size_t)(b * NHEAD + h)) * TNEW * DHEAD;
  const half_t* Kb = khT + ((size_t)(b * NHEAD + h)) * TTOT * DHEAD;
  const half_t* Vb = vhT + ((size_t)(b * NHEAD + h)) * TTOT * DHEAD;

  // Q fragments for DH = 0..31 and 32..63
  v16h qf0, qf1;
  {
    int qrow = t0 + l15;
#pragma unroll
    for (int j = 0; j < 16; ++j) {
      int dd = frag_ak(j, lhalf);
      qf0[j] = Qb[(size_t)qrow * DHEAD + dd];
      qf1[j] = Qb[(size_t)qrow * DHEAD + 32 + dd];
    }
  }

  float m_i[8], l_i[8];
#pragma unroll
  for (int r = 0; r < 8; ++r) { m_i[r] = -1e30f; l_i[r] = 0.f; }
  v8f cacc[4] = {};

  const int nkb = (tlen + 31) >> 5;
  for (int kb = 0; kb < nkb; ++kb) {
    const int kbase = kb * 32;
    const int key0 = kbase + l15, key1 = kbase + 16 + l15;

    v16h kf00, kf01, kf10, kf11;
#pragma unroll
    for (int j = 0; j < 16; ++j) {
      int dd = frag_bk(j, lhalf);
      kf00[j] = Kb[(size_t)key0 * DHEAD + dd];
      kf01[j] = Kb[(size_t)key0 * DHEAD + 32 + dd];
      kf10[j] = Kb[(size_t)key1 * DHEAD + dd];
      kf11[j] = Kb[(size_t)key1 * DHEAD + 32 + dd];
    }
    v8f s0 = {}, s1 = {};
    s0 = wmma16(qf0, kf00, s0);
    s0 = wmma16(qf1, kf01, s0);
    s1 = wmma16(qf0, kf10, s1);
    s1 = wmma16(qf1, kf11, s1);

    // online softmax (rows r+8*lhalf, columns spread over 16-lane halves)
    const float sc = 0.125f;  // 1/sqrt(64)
    float p0[8], p1[8], scale[8];
#pragma unroll
    for (int r = 0; r < 8; ++r) {
      float a = (key0 < tlen) ? s0[r] * sc : -1e30f;
      float c = (key1 < tlen) ? s1[r] * sc : -1e30f;
      float rm = fmaxf(a, c);
#pragma unroll
      for (int mk = 1; mk < 16; mk <<= 1) rm = fmaxf(rm, __shfl_xor(rm, mk));
      float nm = fmaxf(m_i[r], rm);
      float sf = __expf(m_i[r] - nm);
      float e0 = __expf(a - nm), e1 = __expf(c - nm);
      float rs = e0 + e1;
#pragma unroll
      for (int mk = 1; mk < 16; mk <<= 1) rs += __shfl_xor(rs, mk);
      l_i[r] = l_i[r] * sf + rs;
      m_i[r] = nm;
      scale[r] = sf;
      p0[r] = e0;
      p1[r] = e1;
    }
#pragma unroll
    for (int dt = 0; dt < 4; ++dt)
#pragma unroll
      for (int r = 0; r < 8; ++r) cacc[dt][r] *= scale[r];

    // stage P (f16) through wave-private LDS to reshape into an A-fragment
#pragma unroll
    for (int r = 0; r < 8; ++r) {
      int mrow = r + 8 * lhalf;
      Sbuf[wave][mrow][l15]      = (half_t)p0[r];
      Sbuf[wave][mrow][16 + l15] = (half_t)p1[r];
    }
    v16h pf;
#pragma unroll
    for (int j = 0; j < 16; ++j) pf[j] = Sbuf[wave][l15][frag_ak(j, lhalf)];

#pragma unroll
    for (int dt = 0; dt < 4; ++dt) {
      v16h vfrag;
#pragma unroll
      for (int j = 0; j < 16; ++j) {
        int kk = kbase + frag_bk(j, lhalf);
        vfrag[j] = Vb[(size_t)kk * DHEAD + dt * 16 + l15];
      }
      cacc[dt] = wmma16(pf, vfrag, cacc[dt]);
    }
  }

  // normalize and emit ctx in [B, T, H*DH] layout (f16 input for Wo GEMM)
#pragma unroll
  for (int r = 0; r < 8; ++r) {
    float inv = (l_i[r] > 0.f) ? 1.f / l_i[r] : 0.f;
    int mrow = r + 8 * lhalf;
    int t = t0 + mrow;
#pragma unroll
    for (int dt = 0; dt < 4; ++dt) {
      float val = cacc[dt][r] * inv;
      ctxh[((size_t)(b * TNEW + t)) * DMODEL + h * DHEAD + dt * 16 + l15] =
          (half_t)val;
    }
  }
}

// ---------------------------------------------------------------------------
// Host-side orchestration
// ---------------------------------------------------------------------------
extern "C" void kernel_launch(void* const* d_in, const int* in_sizes, int n_in,
                              void* d_out, int out_size, void* d_ws,
                              size_t ws_size, hipStream_t stream) {
  (void)in_sizes; (void)n_in; (void)out_size; (void)ws_size;

  const float* x_new    = (const float*)d_in[0];
  const float* inv_freq = (const float*)d_in[1];
  const float* past_k   = (const float*)d_in[2];
  const float* past_v   = (const float*)d_in[3];
  const float* Wq = (const float*)d_in[4];  const float* bq = (const float*)d_in[5];
  const float* Wk = (const float*)d_in[6];  const float* bk = (const float*)d_in[7];
  const float* Wv = (const float*)d_in[8];  const float* bv = (const float*)d_in[9];
  const float* Wo = (const float*)d_in[10]; const float* bo = (const float*)d_in[11];
  const int* past_len              = (const int*)d_in[13];
  const unsigned char* valid_mask  = (const unsigned char*)d_in[14];

  // outputs: out[8,512,1024] f32, k_total & v_total [8,16,2560,64] f32, lens[8] i32
  const size_t OUT_E = (size_t)MROWS * DMODEL;              // 4,194,304
  const size_t KV_E  = (size_t)BSZ * NHEAD * TTOT * DHEAD;  // 20,971,520
  float* out_p  = (float*)d_out;
  float* kT     = out_p + OUT_E;
  float* vT     = kT + KV_E;
  int*   lens_o = (int*)(vT + KV_E);

  // workspace carve-up
  half_t* xh   = (half_t*)d_ws;
  half_t* wqh  = xh + OUT_E;
  half_t* wkh  = wqh + (size_t)DMODEL * DMODEL;
  half_t* wvh  = wkh + (size_t)DMODEL * DMODEL;
  half_t* woh  = wvh + (size_t)DMODEL * DMODEL;
  float*  qf   = (float*)(woh + (size_t)DMODEL * DMODEL);
  float*  kf   = qf + OUT_E;
  float*  vf   = kf + OUT_E;
  half_t* qh   = (half_t*)(vf + OUT_E);
  half_t* khT  = qh + OUT_E;
  half_t* vhT  = khT + KV_E;
  half_t* ctxh = vhT + KV_E;
  int* lens_ws = (int*)(ctxh + OUT_E);

  // 1) precision staging
  k_f32_to_f16<<<2048, 256, 0, stream>>>(x_new, xh, (long)OUT_E);
  k_f32_to_f16<<<1024, 256, 0, stream>>>(Wq, wqh, (long)DMODEL * DMODEL);
  k_f32_to_f16<<<1024, 256, 0, stream>>>(Wk, wkh, (long)DMODEL * DMODEL);
  k_f32_to_f16<<<1024, 256, 0, stream>>>(Wv, wvh, (long)DMODEL * DMODEL);
  k_f32_to_f16<<<1024, 256, 0, stream>>>(Wo, woh, (long)DMODEL * DMODEL);
  k_lens<<<1, 32, 0, stream>>>(valid_mask, past_len, lens_ws, lens_o);

  // 2) QKV projections (WMMA GEMM, TDM-staged tiles)
  dim3 ggrid(MROWS / 128, DMODEL / 128);
  k_gemm_wT<<<ggrid, 256, 0, stream>>>(xh, wqh, bq, qf, MROWS, DMODEL, DMODEL, nullptr);
  k_gemm_wT<<<ggrid, 256, 0, stream>>>(xh, wkh, bk, kf, MROWS, DMODEL, DMODEL, nullptr);
  k_gemm_wT<<<ggrid, 256, 0, stream>>>(xh, wvh, bv, vf, MROWS, DMODEL, DMODEL, nullptr);

  // 3) cache materialization: past copy + ragged RoPE scatter
  k_init_past<<<8192, 256, 0, stream>>>(past_k, past_v, past_len, kT, vT, khT, vhT);
  k_rope_scatter<<<dim3(TNEW, NHEAD, BSZ), 64, 0, stream>>>(
      qf, kf, vf, inv_freq, past_len, valid_mask, qh, kT, vT, khT, vhT);

  // 4) flash attention (WMMA)
  k_attn<<<BSZ * NHEAD * (TNEW / 64), 128, 0, stream>>>(qh, khT, vhT, lens_ws, ctxh);

  // 5) output projection + valid-token mask (WMMA GEMM)
  k_gemm_wT<<<ggrid, 256, 0, stream>>>(ctxh, woh, bo, out_p, MROWS, DMODEL,
                                       DMODEL, valid_mask);
}
